// Policy_21904333209678
// MI455X (gfx1250) — matrix operand order repaired and checked by reading, not compiled, not run on hardware
//
#include <hip/hip_runtime.h>

// ---------------------------------------------------------------------------
// Types for CDNA5 WMMA (gfx1250, wave32)
// ---------------------------------------------------------------------------
typedef __bf16 bf16_t;
typedef bf16_t v16bf __attribute__((ext_vector_type(16)));
typedef float  v8f   __attribute__((ext_vector_type(8)));

union Frag {
    v16bf v;
    uint4 q[2];
};

__device__ __forceinline__ unsigned short f2bf(float f) {
    union { float f; unsigned u; } v; v.f = f;
    unsigned r = v.u + 0x7FFFu + ((v.u >> 16) & 1u);   // round-to-nearest-even
    return (unsigned short)(r >> 16);
}

__device__ __forceinline__ float elu_f(float x) {
    return x > 0.f ? x : (__expf(x) - 1.f);
}

// Async DMA: global -> LDS, 16 bytes per lane, tracked by ASYNCcnt.
__device__ __forceinline__ void async_b128(const unsigned short* g,
                                           const unsigned short* lds_ptr) {
    unsigned loff = (unsigned)(uintptr_t)lds_ptr;   // low 32 bits of generic = LDS offset
    asm volatile("global_load_async_to_lds_b128 %0, %1, off"
                 :: "v"(loff), "v"(g) : "memory");
}

// ---------------------------------------------------------------------------
// bf16 WMMA GEMM:  C[M,N] = A[M,K] (bf16) * Wt[Npad,K]^T (bf16) + bias
// Block = 128 threads (4 waves). Tile 128(M) x 64(N), K-step 32, double-
// buffered LDS filled by GLOBAL_LOAD_ASYNC_TO_LDS_B128 (ASYNCcnt).
// Each wave: 32x64 = 2 A-frags x 4 B-frags = 8 WMMAs / K-step.
// Last K-step peeled so the steady-state loop body is branch-free.
// Npad must be a multiple of 64 (Wt zero-padded); store guard uses Nout.
// ---------------------------------------------------------------------------
__global__ __launch_bounds__(128)
void policy_gemm_bf16(const unsigned short* __restrict__ A,   // M x K row-major
                      const unsigned short* __restrict__ Wt,  // Npad x K row-major
                      const float* __restrict__ bias,         // Nout
                      const float* __restrict__ addsrc,       // M x ldc or null
                      float* __restrict__ Cf,                 // M x ldc or null
                      unsigned short* __restrict__ Cbf,       // M x ldcbf or null
                      int M, int Nout, int K, int ldc, int ldcbf, int act)
{
    __shared__ __attribute__((aligned(16))) unsigned short As[2][128 * 32];
    __shared__ __attribute__((aligned(16))) unsigned short Bs[2][64 * 32]; // [n][k]

    const int tid  = threadIdx.x;
    const int wave = tid >> 5;
    const int lane = tid & 31;
    const int hh   = lane >> 4;     // half-wave select
    const int ln16 = lane & 15;
    const int m0   = blockIdx.y * 128;
    const int n0   = blockIdx.x * 64;

    // issue one stage of async DMA: 4 chunks for As, 2 for Bs per thread
    auto stage = [&](int buf, int k0) {
        #pragma unroll
        for (int r = 0; r < 4; ++r) {
            int idx = tid + r * 128;            // 0..511
            int row = idx >> 2;                 // 0..127
            int c4  = idx & 3;
            async_b128(A + (size_t)(m0 + row) * K + k0 + c4 * 8,
                       &As[buf][row * 32 + c4 * 8]);
        }
        #pragma unroll
        for (int r = 0; r < 2; ++r) {
            int idx = tid + r * 128;            // 0..255
            int row = idx >> 2;                 // 0..63
            int c4  = idx & 3;
            async_b128(Wt + (size_t)(n0 + row) * K + k0 + c4 * 8,
                       &Bs[buf][row * 32 + c4 * 8]);
        }
    };

    v8f zero = {0.f,0.f,0.f,0.f,0.f,0.f,0.f,0.f};
    v8f acc[2][4];
    #pragma unroll
    for (int i = 0; i < 2; ++i)
        #pragma unroll
        for (int j = 0; j < 4; ++j) acc[i][j] = zero;

    // Load ALL fragments of buffer p first (batched ds_load_b128 under one
    // wait window), then the 8 WMMAs.
    auto compute = [&](int p) {
        Frag af[2], bfr[4];
        const unsigned short* a0 = &As[p][(wave * 32 + ln16) * 32];
        af[0].q[0] = *(const uint4*)(a0 + 8 * hh);
        af[0].q[1] = *(const uint4*)(a0 + 16 + 8 * hh);
        const unsigned short* a1 = a0 + 16 * 32;
        af[1].q[0] = *(const uint4*)(a1 + 8 * hh);
        af[1].q[1] = *(const uint4*)(a1 + 16 + 8 * hh);
        #pragma unroll
        for (int j = 0; j < 4; ++j) {
            const unsigned short* br = &Bs[p][(j * 16 + ln16) * 32];
            bfr[j].q[0] = *(const uint4*)(br + 8 * hh);
            bfr[j].q[1] = *(const uint4*)(br + 16 + 8 * hh);
        }
        #pragma unroll
        for (int i = 0; i < 2; ++i)
            #pragma unroll
            for (int j = 0; j < 4; ++j)
                acc[i][j] = __builtin_amdgcn_wmma_f32_16x16x32_bf16(
                                false, af[i].v, false, bfr[j].v,
                                (short)0, acc[i][j], false, false);
    };

    stage(0, 0);
    int p = 0;
    int k0 = 0;
    // steady state: branch-free body, DMA of next tile overlaps WMMAs
    for (; k0 + 32 < K; k0 += 32) {
        stage(p ^ 1, k0 + 32);
        asm volatile("s_wait_asynccnt 0x6" ::: "memory"); // stage p landed
        __syncthreads();
        compute(p);
        __syncthreads();   // all waves done reading buf p before re-fill
        p ^= 1;
    }
    // peeled final K-step
    asm volatile("s_wait_asynccnt 0x0" ::: "memory");
    __syncthreads();
    compute(p);

    // Epilogue. D layout: VGPR r -> M = r + 8*hh ; N = ln16 (+16j)
    #pragma unroll
    for (int i = 0; i < 2; ++i) {
        #pragma unroll
        for (int j = 0; j < 4; ++j) {
            int n = n0 + j * 16 + ln16;
            if (n >= Nout) continue;
            float bs = bias ? bias[n] : 0.f;
            #pragma unroll
            for (int r = 0; r < 8; ++r) {
                int m = m0 + wave * 32 + i * 16 + hh * 8 + r;
                float v = acc[i][j][r] + bs;
                if (addsrc) v += addsrc[(size_t)m * ldc + n];
                if (act == 1) v = elu_f(v);
                if (Cf)  Cf[(size_t)m * ldc + n] = v;
                if (Cbf) Cbf[(size_t)m * ldcbf + n] = f2bf(v);
            }
        }
    }
}

// ---------------------------------------------------------------------------
// LayerNorm over D=512, optional ELU, optional f32 / bf16 outputs
// ---------------------------------------------------------------------------
__global__ __launch_bounds__(128)
void policy_ln(const float* __restrict__ X, const float* __restrict__ g,
               const float* __restrict__ b, float* __restrict__ outf,
               unsigned short* __restrict__ outbf, int ldbf, int D, int act)
{
    const int row = blockIdx.x;
    const float* x = X + (size_t)row * D;
    float s = 0.f, s2 = 0.f;
    for (int d = threadIdx.x; d < D; d += 128) { float v = x[d]; s += v; s2 += v * v; }
    __shared__ float red[128], red2[128];
    red[threadIdx.x] = s; red2[threadIdx.x] = s2;
    __syncthreads();
    for (int off = 64; off > 0; off >>= 1) {
        if (threadIdx.x < off) {
            red[threadIdx.x]  += red[threadIdx.x + off];
            red2[threadIdx.x] += red2[threadIdx.x + off];
        }
        __syncthreads();
    }
    float mean = red[0] / (float)D;
    float var  = red2[0] / (float)D - mean * mean;
    float inv  = rsqrtf(var + 1e-5f);
    for (int d = threadIdx.x; d < D; d += 128) {
        float v = (x[d] - mean) * inv * g[d] + b[d];
        if (act == 1) v = elu_f(v);
        if (outf)  outf[(size_t)row * D + d] = v;
        if (outbf) outbf[(size_t)row * ldbf + d] = f2bf(v);
    }
}

// ---------------------------------------------------------------------------
// Conv (K=4 depthwise over time) + SiLU; also emits next_conv state
// ---------------------------------------------------------------------------
__global__ __launch_bounds__(256)
void policy_conv(const float* __restrict__ proj,        // B x 2048 (u | z)
                 const float* __restrict__ conv_state,  // B x 4 x 3 x 1024
                 const float* __restrict__ cW,          // 4 x 1024
                 const float* __restrict__ cb,          // 1024
                 float* __restrict__ uf, unsigned short* __restrict__ ubf,
                 float* __restrict__ out_conv, int layer)
{
    int idx = blockIdx.x * 256 + threadIdx.x;  // b*1024 + c
    int c  = idx & 1023;
    int bb = idx >> 10;
    const float* st = conv_state + (((size_t)bb * 4 + layer) * 3) * 1024 + c;
    float s0 = st[0], s1 = st[1024], s2 = st[2048];
    float up = proj[(size_t)bb * 2048 + c];
    float v = s0 * cW[c] + s1 * cW[1024 + c] + s2 * cW[2048 + c]
            + up * cW[3072 + c] + cb[c];
    float sv = v / (1.f + __expf(-v));          // silu
    uf[idx]  = sv;
    ubf[idx] = f2bf(sv);
    float* oc = out_conv + (((size_t)bb * 4 + layer) * 3) * 1024 + c;
    oc[0] = s1; oc[1024] = s2; oc[2048] = up;
}

// ---------------------------------------------------------------------------
// SSM recurrence + gate; one thread per (b, c), STATE=32 vectorized float4
// ---------------------------------------------------------------------------
__global__ __launch_bounds__(256)
void policy_ssm(const float* __restrict__ xp,        // B x 1088
                const float* __restrict__ proj,      // B x 2048 (z at 1024+)
                const float* __restrict__ uf,        // B x 1024
                const float* __restrict__ ssm_state, // B x 4 x 1024 x 32
                const float* __restrict__ A_log,     // 1024 x 32
                const float* __restrict__ Dp,
                const float* __restrict__ dt_bias,
                float* __restrict__ out_ssm, unsigned short* __restrict__ ybf,
                int layer)
{
    int bb = blockIdx.x >> 2;
    int c  = ((blockIdx.x & 3) << 8) + threadIdx.x;
    __shared__ float bt[32], ct[32];
    if (threadIdx.x < 32)       bt[threadIdx.x]      = xp[(size_t)bb * 1088 + 1024 + threadIdx.x];
    else if (threadIdx.x < 64)  ct[threadIdx.x - 32] = xp[(size_t)bb * 1088 + 1056 + (threadIdx.x - 32)];
    __syncthreads();

    float dtr = xp[(size_t)bb * 1088 + c] + dt_bias[c];
    float dt  = dtr > 20.f ? dtr : __logf(1.f + __expf(dtr));   // softplus
    float u   = uf[(size_t)bb * 1024 + c];
    float z   = proj[(size_t)bb * 2048 + 1024 + c];
    float zs  = z / (1.f + __expf(-z));                         // silu(z)

    const float* sold = ssm_state + ((((size_t)bb * 4 + layer) * 1024) + c) * 32;
    float*       snew = out_ssm   + ((((size_t)bb * 4 + layer) * 1024) + c) * 32;
    const float* al   = A_log + (size_t)c * 32;

    float acc = 0.f;
    float dtu = dt * u;
    #pragma unroll
    for (int s4 = 0; s4 < 8; ++s4) {
        float4 so = ((const float4*)sold)[s4];
        float4 av = ((const float4*)al)[s4];
        float4 ns;
        ns.x = __expf(-dt * __expf(av.x)) * so.x + dtu * bt[s4 * 4 + 0];
        ns.y = __expf(-dt * __expf(av.y)) * so.y + dtu * bt[s4 * 4 + 1];
        ns.z = __expf(-dt * __expf(av.z)) * so.z + dtu * bt[s4 * 4 + 2];
        ns.w = __expf(-dt * __expf(av.w)) * so.w + dtu * bt[s4 * 4 + 3];
        ((float4*)snew)[s4] = ns;
        acc += ns.x * ct[s4 * 4] + ns.y * ct[s4 * 4 + 1]
             + ns.z * ct[s4 * 4 + 2] + ns.w * ct[s4 * 4 + 3];
    }
    float y = (acc + Dp[c] * u) * zs;
    ybf[(size_t)bb * 1024 + c] = f2bf(y);
}

// ---------------------------------------------------------------------------
// Weight transpose+convert with zero row padding, f32->bf16 convert, f32 copy
// ---------------------------------------------------------------------------
__global__ void policy_transpose_bf16(const float* __restrict__ W,
                                      unsigned short* __restrict__ Wt,
                                      int K, int N, int Npad)
{
    size_t i = (size_t)blockIdx.x * 256 + threadIdx.x;
    if (i >= (size_t)K * Npad) return;
    int n = (int)(i / (size_t)K);
    int k = (int)(i % (size_t)K);
    Wt[i] = (n < N) ? f2bf(W[(size_t)k * N + n]) : (unsigned short)0;
}

__global__ void policy_cvt_bf16(const float* __restrict__ in,
                                unsigned short* __restrict__ out, int n)
{
    int i = blockIdx.x * 256 + threadIdx.x;
    if (i < n) out[i] = f2bf(in[i]);
}

__global__ void policy_copy_f32(const float* __restrict__ in,
                                float* __restrict__ out, int n)
{
    int i = blockIdx.x * 256 + threadIdx.x;
    if (i < n) out[i] = in[i];
}

// ---------------------------------------------------------------------------
// Host orchestration
// ---------------------------------------------------------------------------
extern "C" void kernel_launch(void* const* d_in, const int* in_sizes, int n_in,
                              void* d_out, int out_size, void* d_ws, size_t ws_size,
                              hipStream_t stream)
{
    (void)in_sizes; (void)n_in; (void)out_size; (void)ws_size;
    const int B = 512, D = 512, INNER = 1024, STATE = 32, L = 4, OBS = 64;

    // ---- inputs (setup_inputs dict insertion order) ----
    const float* obs        = (const float*)d_in[0];
    const float* ssm_state  = (const float*)d_in[1];
    const float* conv_state = (const float*)d_in[2];
    auto LP = [&](int l, int j) { return (const float*)d_in[3 + l * 13 + j]; };
    // per-layer j: 0 norm_g 1 norm_b 2 in_W 3 in_b 4 x_W 5 x_b 6 out_W 7 out_b
    //             8 conv_W 9 conv_b 10 A_log 11 Dp 12 dt_bias
    const float* menc_W  = (const float*)d_in[55];
    const float* menc_b  = (const float*)d_in[56];
    const float* menc_g  = (const float*)d_in[57];
    const float* menc_lb = (const float*)d_in[58];
    const float* oenc_W  = (const float*)d_in[59];
    const float* oenc_b  = (const float*)d_in[60];
    const float* oenc_g  = (const float*)d_in[61];
    const float* oenc_lb = (const float*)d_in[62];
    const float* out_g   = (const float*)d_in[63];
    const float* out_b   = (const float*)d_in[64];
    const float* down_W  = (const float*)d_in[65];
    const float* down_b  = (const float*)d_in[66];
    const float* t1_W    = (const float*)d_in[67];
    const float* t1_b    = (const float*)d_in[68];
    const float* t1_g    = (const float*)d_in[69];
    const float* t1_lb   = (const float*)d_in[70];
    const float* t2_W    = (const float*)d_in[71];
    const float* t2_b    = (const float*)d_in[72];
    const float* t3_W    = (const float*)d_in[73];
    const float* t3_b    = (const float*)d_in[74];
    const float* mh_W    = (const float*)d_in[75];
    const float* mh_b    = (const float*)d_in[76];
    const float* logstd  = (const float*)d_in[77];

    // ---- outputs ----
    float* out_mean   = (float*)d_out;                           // 512 x 32
    float* out_logstd = out_mean + 16384;                        // 32
    float* out_ssm    = out_logstd + 32;                         // 512x4x1024x32
    float* out_conv   = out_ssm + (size_t)B * L * INNER * STATE; // 512x4x3x1024

    // ---- workspace carve ----
    char* wsp = (char*)d_ws;
    size_t off = 0;
    auto alloc = [&](size_t bytes) -> void* {
        void* p = wsp + off; off += (bytes + 255) & ~(size_t)255; return p;
    };
    unsigned short* obs_bf = (unsigned short*)alloc((size_t)B * OBS * 2);
    float*          x      = (float*)alloc((size_t)B * D * 4);
    float*          ench   = (float*)alloc((size_t)B * D * 4);
    unsigned short* h_bf   = (unsigned short*)alloc((size_t)B * D * 2);
    float*          proj   = (float*)alloc((size_t)B * 2048 * 4);
    float*          uf     = (float*)alloc((size_t)B * INNER * 4);
    unsigned short* u_bf   = (unsigned short*)alloc((size_t)B * INNER * 2);
    float*          xp     = (float*)alloc((size_t)B * 1088 * 4);
    unsigned short* y_bf   = (unsigned short*)alloc((size_t)B * INNER * 2);
    unsigned short* t_bf   = (unsigned short*)alloc((size_t)B * 768 * 2);
    unsigned short* m_bf   = (unsigned short*)alloc((size_t)B * D * 2);
    float*          h1f    = (float*)alloc((size_t)B * 512 * 4);
    unsigned short* h1_bf  = (unsigned short*)alloc((size_t)B * 512 * 2);
    unsigned short* h2_bf  = (unsigned short*)alloc((size_t)B * 256 * 2);
    unsigned short* h3_bf  = (unsigned short*)alloc((size_t)B * 128 * 2);

    auto allocW = [&](size_t elems) { return (unsigned short*)alloc(elems * 2); };
    unsigned short *in_Wt[4], *x_Wt[4], *out_Wt[4];
    for (int l = 0; l < 4; ++l) {
        in_Wt[l]  = allocW((size_t)512 * 2048);
        x_Wt[l]   = allocW((size_t)1024 * 1088);
        out_Wt[l] = allocW((size_t)1024 * 512);
    }
    unsigned short* menc_Wt = allocW((size_t)64 * 512);
    unsigned short* oenc_Wt = allocW((size_t)64 * 512);
    unsigned short* down_Wt = allocW((size_t)512 * 256);
    unsigned short* t1_Wt   = allocW((size_t)768 * 512);
    unsigned short* t2_Wt   = allocW((size_t)512 * 256);
    unsigned short* t3_Wt   = allocW((size_t)256 * 128);
    unsigned short* mh_Wt   = allocW((size_t)128 * 64);   // padded to 64 rows

    auto transpose = [&](const float* W, unsigned short* Wt, int K, int N, int Npad) {
        size_t tot = (size_t)K * Npad;
        policy_transpose_bf16<<<(int)((tot + 255) / 256), 256, 0, stream>>>(W, Wt, K, N, Npad);
    };
    for (int l = 0; l < 4; ++l) {
        transpose(LP(l, 2), in_Wt[l], 512, 2048, 2048);
        transpose(LP(l, 4), x_Wt[l], 1024, 1088, 1088);
        transpose(LP(l, 6), out_Wt[l], 1024, 512, 512);
    }
    transpose(menc_W, menc_Wt, 64, 512, 512);
    transpose(oenc_W, oenc_Wt, 64, 512, 512);
    transpose(down_W, down_Wt, 512, 256, 256);
    transpose(t1_W, t1_Wt, 768, 512, 512);
    transpose(t2_W, t2_Wt, 512, 256, 256);
    transpose(t3_W, t3_Wt, 256, 128, 128);
    transpose(mh_W, mh_Wt, 128, 32, 64);

    policy_cvt_bf16<<<(B * OBS + 255) / 256, 256, 0, stream>>>(obs, obs_bf, B * OBS);

    auto gemm = [&](const unsigned short* A, const unsigned short* Wt, const float* bias,
                    const float* addsrc, float* Cf, unsigned short* Cbf,
                    int Npad, int Nout, int K, int ldc, int ldcbf, int act) {
        dim3 grid(Npad / 64, B / 128);
        policy_gemm_bf16<<<grid, 128, 0, stream>>>(A, Wt, bias, addsrc, Cf, Cbf,
                                                   B, Nout, K, ldc, ldcbf, act);
    };
    auto ln = [&](const float* X, const float* g, const float* b,
                  float* outf, unsigned short* outbf, int ldbf, int act) {
        policy_ln<<<B, 128, 0, stream>>>(X, g, b, outf, outbf, ldbf, 512, act);
    };

    // ---- motion encoder: x = elu(LN(obs @ menc_W + b)) ----
    gemm(obs_bf, menc_Wt, menc_b, nullptr, ench, nullptr, 512, 512, 64, 512, 0, 0);
    ln(ench, menc_g, menc_lb, x, nullptr, 0, 1);

    // ---- 4 SSM layers ----
    for (int l = 0; l < 4; ++l) {
        ln(x, LP(l, 0), LP(l, 1), nullptr, h_bf, 512, 0);                 // h = LN(x)
        gemm(h_bf, in_Wt[l], LP(l, 3), nullptr, proj, nullptr, 2048, 2048, 512, 2048, 0, 0);
        policy_conv<<<(B * INNER) / 256, 256, 0, stream>>>(
            proj, conv_state, LP(l, 8), LP(l, 9), uf, u_bf, out_conv, l);
        gemm(u_bf, x_Wt[l], LP(l, 5), nullptr, xp, nullptr, 1088, 1088, 1024, 1088, 0, 0);
        policy_ssm<<<B * 4, 256, 0, stream>>>(
            xp, proj, uf, ssm_state, LP(l, 10), LP(l, 11), LP(l, 12),
            out_ssm, y_bf, l);
        // x = x + y @ out_W + out_b   (residual fused, in-place)
        gemm(y_bf, out_Wt[l], LP(l, 7), x, x, nullptr, 512, 512, 1024, 512, 0, 0);
    }

    // ---- obs encoder -> t_bf[:, 0:512] ----
    gemm(obs_bf, oenc_Wt, oenc_b, nullptr, ench, nullptr, 512, 512, 64, 512, 0, 0);
    ln(ench, oenc_g, oenc_lb, nullptr, t_bf, 768, 1);

    // ---- m = elu(LN(x)); m2 = elu(m @ down_W + b) -> t_bf[:, 512:768] ----
    ln(x, out_g, out_b, nullptr, m_bf, 512, 1);
    gemm(m_bf, down_Wt, down_b, nullptr, nullptr, t_bf + 512, 256, 256, 512, 256, 768, 1);

    // ---- trunk ----
    gemm(t_bf, t1_Wt, t1_b, nullptr, h1f, nullptr, 512, 512, 768, 512, 0, 0);
    ln(h1f, t1_g, t1_lb, nullptr, h1_bf, 512, 1);
    gemm(h1_bf, t2_Wt, t2_b, nullptr, nullptr, h2_bf, 256, 256, 512, 256, 256, 1);
    gemm(h2_bf, t3_Wt, t3_b, nullptr, nullptr, h3_bf, 128, 128, 256, 128, 128, 1);
    gemm(h3_bf, mh_Wt, mh_b, nullptr, out_mean, nullptr, 64, 32, 128, 32, 0, 0);

    policy_copy_f32<<<1, 32, 0, stream>>>(logstd, out_logstd, 32);
}
